// MultiHeadAttention_1357209666163
// MI455X (gfx1250) — matrix operand order repaired and checked by reading, not compiled, not run on hardware
//
#include <hip/hip_runtime.h>
#include <stdint.h>

// ---------------------------------------------------------------------------
// MI455X / gfx1250: wave32, WMMA 16x16x32 bf16 (f32 accum), async LDS copies.
// ---------------------------------------------------------------------------

#define BB   4
#define SS   2048
#define DD   1024
#define HH   16
#define DKV  64

typedef __attribute__((ext_vector_type(16))) __bf16 v16bf;
typedef __attribute__((ext_vector_type(8)))  float  v8f;

#define USE_ASYNC_LDS 1

__device__ __forceinline__ v8f wmma_bf16(v16bf a, v16bf b, v8f c) {
  // 8 args: (neg_a, A, neg_b, B, c_mod, C, reuse_a, reuse_b)
  return __builtin_amdgcn_wmma_f32_16x16x32_bf16(false, a, false, b, (short)0, c,
                                                 false, false);
}

// A-matrix 16x32 bf16 fragment: lanes 0-15 M=0..15 K{0-15},
// lanes 16-31 same M, K{16-31}. Pairs contiguous along K.
__device__ __forceinline__ v16bf load_a_frag(const __bf16* base, int lda, int lane) {
  const int half = lane >> 4;
  const __bf16* row = base + (lane & 15) * lda;
  v16bf a;
#pragma unroll
  for (int j = 0; j < 8; ++j) {
    const int kb = 2 * j + half * 16;
    a[2 * j]     = row[kb];
    a[2 * j + 1] = row[kb + 1];
  }
  return a;
}

// B-matrix 32x16 bf16 fragment from a TRANSPOSED tile (tile stored [n][k]):
// lanes 0-15: N=lane, K=2j,2j+1 ; lanes 16-31: N=lane-16, K=16+2j,16+2j+1.
__device__ __forceinline__ v16bf load_bT_frag(const __bf16* tileT, int n0, int k0,
                                              int ldk, int lane) {
  const int half = lane >> 4;
  const __bf16* row = tileT + (n0 + (lane & 15)) * ldk + k0 + half * 16;
  v16bf b;
#pragma unroll
  for (int j = 0; j < 8; ++j) {
    b[2 * j]     = row[2 * j];
    b[2 * j + 1] = row[2 * j + 1];
  }
  return b;
}

// B-matrix 32x16 bf16 fragment from a NATURAL [k][n] tile (strided).
__device__ __forceinline__ v16bf load_b_frag_nat(const __bf16* tile, int n0,
                                                 int ldn, int lane) {
  const int half = lane >> 4;
  const int n = n0 + (lane & 15);
  v16bf b;
#pragma unroll
  for (int j = 0; j < 8; ++j) {
    const int kk = half * 16 + 2 * j;
    b[2 * j]     = tile[kk * ldn + n];
    b[2 * j + 1] = tile[(kk + 1) * ldn + n];
  }
  return b;
}

__device__ __forceinline__ void async_copy_b128(void* lds_dst, const void* gsrc) {
#if USE_ASYNC_LDS
  const uint32_t loff = (uint32_t)(uintptr_t)lds_dst;   // low 32 bits = LDS addr
  asm volatile("global_load_async_to_lds_b128 %0, %1, off"
               :: "v"(loff), "v"(gsrc) : "memory");
#else
  *(uint4*)lds_dst = *(const uint4*)gsrc;
#endif
}

__device__ __forceinline__ void wait_async_copies() {
#if USE_ASYNC_LDS
  asm volatile("s_wait_asynccnt 0x0" ::: "memory");
#endif
}

// ---------------------------------------------------------------------------
// Kernel 1: QKV projection.  C = bf16(X[8192x1024]) @ bf16(W[1024x1024]) + b,
// head-split bf16 store into (B,H,S,64).  blockIdx.z selects q / k / v.
// Block tile 128x128, 8 waves of 32x64, K-step 32.
// ---------------------------------------------------------------------------
__global__ __launch_bounds__(256) void gemm_qkv_kernel(
    const float* __restrict__ q, const float* __restrict__ k,
    const float* __restrict__ v,
    const float* __restrict__ wq, const float* __restrict__ bq,
    const float* __restrict__ wk, const float* __restrict__ bk,
    const float* __restrict__ wv, const float* __restrict__ bv,
    __bf16* __restrict__ qh, __bf16* __restrict__ kh, __bf16* __restrict__ vh) {
  const int z = blockIdx.z;
  const float* X    = (z == 0) ? q  : (z == 1) ? k  : v;
  const float* W    = (z == 0) ? wq : (z == 1) ? wk : wv;
  const float* bias = (z == 0) ? bq : (z == 1) ? bk : bv;
  __bf16* dst       = (z == 0) ? qh : (z == 1) ? kh : vh;

  __shared__ __bf16 As[128 * 34];   // [m][k], stride 34 (pad)
  __shared__ __bf16 Bs[128 * 34];   // [n][k]  (W transposed), stride 34

  const int tid  = threadIdx.x;
  const int lane = tid & 31;
  const int wave = tid >> 5;
  const int wm = (wave >> 1) * 32;        // 0,32,64,96
  const int wn = (wave & 1) * 64;         // 0,64
  const int m_block = blockIdx.y * 128;
  const int n_block = blockIdx.x * 128;

  v8f acc[2][4];
#pragma unroll
  for (int s = 0; s < 2; ++s)
#pragma unroll
    for (int t = 0; t < 4; ++t) acc[s][t] = (v8f){0, 0, 0, 0, 0, 0, 0, 0};

  for (int k0 = 0; k0 < DD; k0 += 32) {
    // A tile 128x32 f32 -> bf16
#pragma unroll
    for (int p = 0; p < 4; ++p) {
      const int idx = tid + p * 256;            // 0..1023
      const int r = idx >> 3, c4 = (idx & 7) * 4;
      const float4 f =
          *(const float4*)(X + (size_t)(m_block + r) * DD + k0 + c4);
      __bf16* d = &As[r * 34 + c4];
      d[0] = (__bf16)f.x; d[1] = (__bf16)f.y;
      d[2] = (__bf16)f.z; d[3] = (__bf16)f.w;
    }
    // W tile 32x128 f32 -> bf16, stored transposed [n][k]
#pragma unroll
    for (int p = 0; p < 4; ++p) {
      const int idx = tid + p * 256;
      const int kk = idx >> 5;                  // 0..31
      const int c4 = (idx & 31) * 4;            // 0..124
      const float4 f =
          *(const float4*)(W + (size_t)(k0 + kk) * DD + n_block + c4);
      Bs[(c4 + 0) * 34 + kk] = (__bf16)f.x;
      Bs[(c4 + 1) * 34 + kk] = (__bf16)f.y;
      Bs[(c4 + 2) * 34 + kk] = (__bf16)f.z;
      Bs[(c4 + 3) * 34 + kk] = (__bf16)f.w;
    }
    __syncthreads();

    const v16bf a0 = load_a_frag(&As[(wm + 0) * 34], 34, lane);
    const v16bf a1 = load_a_frag(&As[(wm + 16) * 34], 34, lane);
#pragma unroll
    for (int t = 0; t < 4; ++t) {
      const v16bf b = load_bT_frag(Bs, wn + t * 16, 0, 34, lane);
      acc[0][t] = wmma_bf16(a0, b, acc[0][t]);
      acc[1][t] = wmma_bf16(a1, b, acc[1][t]);
    }
    __syncthreads();
  }

  const int half = lane >> 4, nl = lane & 15;
#pragma unroll
  for (int s = 0; s < 2; ++s)
#pragma unroll
    for (int t = 0; t < 4; ++t) {
      const int n_glob = n_block + wn + t * 16 + nl;
      const float bn = bias[n_glob];
      const int h = n_glob >> 6, dk = n_glob & 63;
#pragma unroll
      for (int r = 0; r < 8; ++r) {
        const int m_glob = m_block + wm + s * 16 + r + 8 * half;
        const int bb = m_glob >> 11, srow = m_glob & (SS - 1);
        dst[(((size_t)bb * HH + h) * SS + srow) * DKV + dk] =
            (__bf16)(acc[s][t][r] + bn);
      }
    }
}

// ---------------------------------------------------------------------------
// Kernel 2: flash attention. grid=(S/128, H, B), 8 waves x 16 query rows.
// K/V tiles (32x64 bf16, contiguous per head) streamed via async LDS copies.
// ---------------------------------------------------------------------------
__global__ __launch_bounds__(256) void attention_kernel(
    const __bf16* __restrict__ qh, const __bf16* __restrict__ kh,
    const __bf16* __restrict__ vh, const int* __restrict__ mask,
    __bf16* __restrict__ aout) {
  const int b = blockIdx.z, h = blockIdx.y;
  const int tid = threadIdx.x, lane = tid & 31, wave = tid >> 5;
  const int q0 = blockIdx.x * 128 + wave * 16;
  const size_t headbase = ((size_t)b * HH + h) * SS * DKV;

  __shared__ __bf16 Kt[32 * 64];       // [kv][dk]  (natural => B^T for QK^T)
  __shared__ __bf16 Vt[32 * 64];       // [kv][dv]
  __shared__ __bf16 Pt[8][16 * 32];    // per-wave P staging (C-layout -> A-layout)

  const int half = lane >> 4, nl = lane & 15;
  const float scale = 0.125f;          // 1/sqrt(64)
  const float L2E = 1.44269504f;

  const v16bf qa0 = load_a_frag(qh + headbase + (size_t)q0 * DKV, DKV, lane);
  const v16bf qa1 = load_a_frag(qh + headbase + (size_t)q0 * DKV + 32, DKV, lane);

  v8f o[4];
#pragma unroll
  for (int t = 0; t < 4; ++t) o[t] = (v8f){0, 0, 0, 0, 0, 0, 0, 0};
  float mi[8], li[8];
#pragma unroll
  for (int r = 0; r < 8; ++r) { mi[r] = -3.0e38f; li[r] = 0.0f; }

  for (int it = 0; it < SS / 32; ++it) {
    const int kv0 = it * 32;
    __syncthreads();
    {
      const __bf16* ksrc = kh + headbase + (size_t)kv0 * DKV;
      const __bf16* vsrc = vh + headbase + (size_t)kv0 * DKV;
      async_copy_b128(&Kt[tid * 8], ksrc + tid * 8);   // 256 x 16B = 4KB
      async_copy_b128(&Vt[tid * 8], vsrc + tid * 8);
      wait_async_copies();
    }
    __syncthreads();

    // Scores: S[16x32] = Q[16x64] @ K^T, two 16x16 C tiles, K-dim chained.
    const v16bf kb00 = load_bT_frag(Kt, 0, 0, 64, lane);
    const v16bf kb01 = load_bT_frag(Kt, 0, 32, 64, lane);
    const v16bf kb10 = load_bT_frag(Kt, 16, 0, 64, lane);
    const v16bf kb11 = load_bT_frag(Kt, 16, 32, 64, lane);
    const v8f zc = (v8f){0, 0, 0, 0, 0, 0, 0, 0};
    v8f s0 = wmma_bf16(qa0, kb00, zc); s0 = wmma_bf16(qa1, kb01, s0);
    v8f s1 = wmma_bf16(qa0, kb10, zc); s1 = wmma_bf16(qa1, kb11, s1);

    // Online softmax per row (row = lanes{half*16..+15} at C-reg r).
#pragma unroll
    for (int r = 0; r < 8; ++r) {
      const int qrow = q0 + r + 8 * half;
      const int* mp = mask + ((size_t)b * SS + qrow) * SS + kv0;
      const float x0 = mp[nl]      ? s0[r] * scale : -1.0e9f;
      const float x1 = mp[16 + nl] ? s1[r] * scale : -1.0e9f;
      float vm = fmaxf(x0, x1);
      vm = fmaxf(vm, __shfl_xor(vm, 1, 32));
      vm = fmaxf(vm, __shfl_xor(vm, 2, 32));
      vm = fmaxf(vm, __shfl_xor(vm, 4, 32));
      vm = fmaxf(vm, __shfl_xor(vm, 8, 32));
      const float mnew = fmaxf(mi[r], vm);
      const float p0 = exp2f((x0 - mnew) * L2E);
      const float p1 = exp2f((x1 - mnew) * L2E);
      float sum = p0 + p1;
      sum += __shfl_xor(sum, 1, 32);
      sum += __shfl_xor(sum, 2, 32);
      sum += __shfl_xor(sum, 4, 32);
      sum += __shfl_xor(sum, 8, 32);
      const float fct = exp2f((mi[r] - mnew) * L2E);
      li[r] = li[r] * fct + sum;
      mi[r] = mnew;
#pragma unroll
      for (int t = 0; t < 4; ++t) o[t][r] *= fct;
      __bf16* prow = &Pt[wave][(r + 8 * half) * 32];
      prow[nl]      = (__bf16)p0;
      prow[16 + nl] = (__bf16)p1;
    }

    // P (16x32) @ V (32x64) accumulate. Per-wave Pt buffer: LDS same-wave
    // ops are in-order, no barrier needed.
    const v16bf pf = load_a_frag(&Pt[wave][0], 32, lane);
#pragma unroll
    for (int t = 0; t < 4; ++t) {
      const v16bf vb = load_b_frag_nat(Vt, t * 16, 64, lane);
      o[t] = wmma_bf16(pf, vb, o[t]);
    }
  }

  float inv[8];
#pragma unroll
  for (int r = 0; r < 8; ++r) inv[r] = 1.0f / li[r];
#pragma unroll
  for (int t = 0; t < 4; ++t) {
    const int col = h * DKV + t * 16 + nl;
#pragma unroll
    for (int r = 0; r < 8; ++r) {
      const int srow = q0 + r + 8 * half;
      aout[((size_t)b * SS + srow) * (HH * DKV) + col] =
          (__bf16)(o[t][r] * inv[r]);
    }
  }
}

// ---------------------------------------------------------------------------
// Kernel 3: output projection.  out = A[8192x1024 bf16] @ bf16(w_o) + b_o (f32)
// ---------------------------------------------------------------------------
__global__ __launch_bounds__(256) void gemm_out_kernel(
    const __bf16* __restrict__ A, const float* __restrict__ W,
    const float* __restrict__ bias, float* __restrict__ out) {
  __shared__ __bf16 As[128 * 34];
  __shared__ __bf16 Bs[128 * 34];

  const int tid = threadIdx.x, lane = tid & 31, wave = tid >> 5;
  const int wm = (wave >> 1) * 32;
  const int wn = (wave & 1) * 64;
  const int m_block = blockIdx.y * 128;
  const int n_block = blockIdx.x * 128;

  v8f acc[2][4];
#pragma unroll
  for (int s = 0; s < 2; ++s)
#pragma unroll
    for (int t = 0; t < 4; ++t) acc[s][t] = (v8f){0, 0, 0, 0, 0, 0, 0, 0};

  for (int k0 = 0; k0 < DD; k0 += 32) {
    // A tile 128x32 bf16 (raw copy, 8 elems per uint4 half)
#pragma unroll
    for (int p = 0; p < 2; ++p) {
      const int idx = tid + p * 256;           // 0..511
      const int r = idx >> 2, g = idx & 3;
      const uint4 u =
          *(const uint4*)(A + (size_t)(m_block + r) * DD + k0 + g * 8);
      uint32_t* d32 = (uint32_t*)&As[r * 34 + g * 8];
      d32[0] = u.x; d32[1] = u.y; d32[2] = u.z; d32[3] = u.w;
    }
    // W tile 32x128 f32 -> bf16 transposed
#pragma unroll
    for (int p = 0; p < 4; ++p) {
      const int idx = tid + p * 256;
      const int kk = idx >> 5;
      const int c4 = (idx & 31) * 4;
      const float4 f =
          *(const float4*)(W + (size_t)(k0 + kk) * DD + n_block + c4);
      Bs[(c4 + 0) * 34 + kk] = (__bf16)f.x;
      Bs[(c4 + 1) * 34 + kk] = (__bf16)f.y;
      Bs[(c4 + 2) * 34 + kk] = (__bf16)f.z;
      Bs[(c4 + 3) * 34 + kk] = (__bf16)f.w;
    }
    __syncthreads();

    const v16bf a0 = load_a_frag(&As[(wm + 0) * 34], 34, lane);
    const v16bf a1 = load_a_frag(&As[(wm + 16) * 34], 34, lane);
#pragma unroll
    for (int t = 0; t < 4; ++t) {
      const v16bf bfr = load_bT_frag(Bs, wn + t * 16, 0, 34, lane);
      acc[0][t] = wmma_bf16(a0, bfr, acc[0][t]);
      acc[1][t] = wmma_bf16(a1, bfr, acc[1][t]);
    }
    __syncthreads();
  }

  const int half = lane >> 4, nl = lane & 15;
#pragma unroll
  for (int s = 0; s < 2; ++s)
#pragma unroll
    for (int t = 0; t < 4; ++t) {
      const int n_glob = n_block + wn + t * 16 + nl;
      const float bn = bias[n_glob];
#pragma unroll
      for (int r = 0; r < 8; ++r) {
        const int m_glob = m_block + wm + s * 16 + r + 8 * half;
        out[(size_t)m_glob * DD + n_glob] = acc[s][t][r] + bn;
      }
    }
}

// ---------------------------------------------------------------------------
extern "C" void kernel_launch(void* const* d_in, const int* in_sizes, int n_in,
                              void* d_out, int out_size, void* d_ws,
                              size_t ws_size, hipStream_t stream) {
  const float* q    = (const float*)d_in[0];
  const float* k    = (const float*)d_in[1];
  const float* v    = (const float*)d_in[2];
  const int*   mask = (const int*)d_in[3];
  const float* w_q  = (const float*)d_in[4];
  const float* b_q  = (const float*)d_in[5];
  const float* w_k  = (const float*)d_in[6];
  const float* b_k  = (const float*)d_in[7];
  const float* w_v  = (const float*)d_in[8];
  const float* b_v  = (const float*)d_in[9];
  const float* w_o  = (const float*)d_in[10];
  const float* b_o  = (const float*)d_in[11];

  const size_t per = (size_t)BB * HH * SS * DKV;  // 8,388,608 bf16 = 16 MB
  __bf16* qh   = (__bf16*)d_ws;
  __bf16* kh   = qh + per;
  __bf16* vh   = kh + per;
  __bf16* aout = vh + per;                        // (B,S,H*DV) bf16

  const dim3 blk(256);
  gemm_qkv_kernel<<<dim3(DD / 128, (BB * SS) / 128, 3), blk, 0, stream>>>(
      q, k, v, w_q, b_q, w_k, b_k, w_v, b_v, qh, kh, vh);
  attention_kernel<<<dim3(SS / 128, HH, BB), blk, 0, stream>>>(qh, kh, vh, mask,
                                                               aout);
  gemm_out_kernel<<<dim3(DD / 128, (BB * SS) / 128, 1), blk, 0, stream>>>(
      aout, w_o, b_o, (float*)d_out);
}
